// AxisMDTA_74045236183623
// MI455X (gfx1250) — matrix operand order repaired and checked by reading, not compiled, not run on hardware
//
#include <hip/hip_runtime.h>

// =====================================================================
// AxisMDTA block for MI455X (gfx1250, wave32, WMMA + TDM).
// Shapes: B=4, C=256, T=64, F=256, H=8, d=32, HID=512. Rows M = B*T*F = 65536.
// All matmuls via v_wmma_f32_16x16x32_bf16 (bf16 A/B, f32 accum).
// Workspace layout (bytes), peak 288 MB with region reuse:
//   [0,64M)     xs f32 (residual)    -> later reused as y2 f32
//   [64M,96M)   xn bf16              -> later reused as attn_out bf16
//   [96M,192M)  qkv bf16             -> later reused as y f32 + yn bf16
//   [192M,288M) q/k/v heads bf16     -> later reused as h1 bf16
// =====================================================================

typedef __attribute__((ext_vector_type(16))) __bf16 bf16x16;
typedef __attribute__((ext_vector_type(8)))  __bf16 bf16x8;
typedef __attribute__((ext_vector_type(8)))  float  f32x8;
typedef __attribute__((ext_vector_type(4)))  unsigned int u32x4;
typedef __attribute__((ext_vector_type(4)))  int i32x4;
typedef __attribute__((ext_vector_type(8)))  int i32x8;

#if defined(__has_builtin)
#if __has_builtin(__builtin_amdgcn_tensor_load_to_lds)
#define HAVE_TDM 1
#endif
#endif
#ifndef HAVE_TDM
#define HAVE_TDM 0
#endif

__device__ __forceinline__ unsigned short f2bf(float f) {
  union { float f; unsigned u; } v; v.f = f;
  unsigned r = v.u + 0x7FFFu + ((v.u >> 16) & 1u);   // round-to-nearest-even
  return (unsigned short)(r >> 16);
}
__device__ __forceinline__ float bf2f(unsigned short h) {
  union { unsigned u; float f; } v; v.u = ((unsigned)h) << 16;
  return v.f;
}

// Build a 16-element bf16 fragment from two contiguous 8-element (16B) runs.
__device__ __forceinline__ bf16x16 load_frag(const unsigned short* p0,
                                             const unsigned short* p1) {
  bf16x8 lo = *(const bf16x8*)p0;
  bf16x8 hi = *(const bf16x8*)p1;
  bf16x16 r;
#pragma unroll
  for (int i = 0; i < 8; ++i) { r[i] = lo[i]; r[i + 8] = hi[i]; }
  return r;
}

__device__ __forceinline__ f32x8 wmma_bf16(bf16x16 a, bf16x16 b, f32x8 c) {
  return __builtin_amdgcn_wmma_f32_16x16x32_bf16(false, a, false, b,
                                                 (short)0, c, false, false);
}

#if HAVE_TDM
// 1-D TDM load: nelem 2-byte elements, contiguous in global, into LDS.
// D# layout per CDNA5 ISA ch.8: group0 = {count, lds_addr, global_addr, type=2},
// group1 = {data_size=2B, tensor_dim0=tile_dim0=nelem, dims1=1, stride0=nelem}.
__device__ __forceinline__ void tdm_load_b16(unsigned lds_addr, const void* gptr,
                                             unsigned nelem) {
  unsigned long long ga = (unsigned long long)(size_t)gptr;
  u32x4 g0;
  g0[0] = 1u;                                   // count=1, user descriptor
  g0[1] = lds_addr;                             // LDS byte address
  g0[2] = (unsigned)(ga & 0xFFFFFFFFu);         // global_addr[31:0]
  g0[3] = (unsigned)((ga >> 32) & 0x01FFFFFFu)  // global_addr[56:32]
          | (2u << 30);                         // type=2 ("image")
  i32x8 g1 = {};
  g1[0] = (int)(1u << 16);                      // wg_mask=0, data_size=1 (2B)
  g1[1] = (int)((nelem & 0xFFFFu) << 16);       // tensor_dim0[15:0] @bits63:48
  g1[2] = (int)(((nelem >> 16) & 0xFFFFu)       // tensor_dim0[31:16]
                | (1u << 16));                  // tensor_dim1=1
  g1[3] = (int)((nelem & 0xFFFFu) << 16);       // tile_dim0 @bits127:112
  g1[4] = 1;                                    // tile_dim1=1, tile_dim2=0
  g1[5] = (int)nelem;                           // tensor_dim0_stride[31:0]
  g1[6] = 0;                                    // stride0 hi, stride1 lo
  g1[7] = 0;
  i32x4 z4 = {};
#if defined(__clang_major__) && (__clang_major__ >= 23)
  i32x8 z8 = {};
  __builtin_amdgcn_tensor_load_to_lds(g0, g1, z4, z4, z8, 0);
#else
  __builtin_amdgcn_tensor_load_to_lds(g0, g1, z4, z4, 0);
#endif
}
#endif

// ---------------------------------------------------------------------
// Kernel 1: pack x (B,C,T,F) f32 -> xs[(b*T+t)*F + f][c] f32 (LDS transpose)
// grid = B*T * (C/32) * (F/32) = 16384, block = 256
// ---------------------------------------------------------------------
__global__ __launch_bounds__(256)
void pack_xs_kernel(const float* __restrict__ x, float* __restrict__ xs) {
  int blk = blockIdx.x;
  int ftile = blk & 7, ctile = (blk >> 3) & 7, bt = blk >> 6;
  int b = bt >> 6, t = bt & 63;
  __shared__ float tile[32][33];
  int ti = threadIdx.x & 31, to = threadIdx.x >> 5;
#pragma unroll
  for (int k = 0; k < 4; ++k) {
    int cc = to + 8 * k;
    tile[cc][ti] =
        x[(((size_t)(b * 256 + ctile * 32 + cc)) * 64 + t) * 256 + ftile * 32 + ti];
  }
  __syncthreads();
#pragma unroll
  for (int k = 0; k < 4; ++k) {
    int ff = to + 8 * k;
    xs[((size_t)bt * 256 + ftile * 32 + ff) * 256 + ctile * 32 + ti] = tile[ti][ff];
  }
}

// ---------------------------------------------------------------------
// LayerNorm over C=256 per row; one wave per row; bf16 output.
// ---------------------------------------------------------------------
__global__ __launch_bounds__(256)
void layernorm_kernel(const float* __restrict__ X, const float* __restrict__ g,
                      const float* __restrict__ be, unsigned short* __restrict__ Y) {
  int warp = threadIdx.x >> 5, lane = threadIdx.x & 31;
  size_t row = (size_t)blockIdx.x * 8 + warp;
  const float* xr = X + row * 256;
  float vals[8];
  float s = 0.f;
#pragma unroll
  for (int j = 0; j < 8; ++j) { vals[j] = xr[lane + 32 * j]; s += vals[j]; }
#pragma unroll
  for (int m = 16; m >= 1; m >>= 1) s += __shfl_xor(s, m, 32);
  float mu = s * (1.f / 256.f);
  float vs = 0.f;
#pragma unroll
  for (int j = 0; j < 8; ++j) { float d = vals[j] - mu; vs += d * d; }
#pragma unroll
  for (int m = 16; m >= 1; m >>= 1) vs += __shfl_xor(vs, m, 32);
  float rsig = rsqrtf(vs * (1.f / 256.f) + 1e-5f);
#pragma unroll
  for (int j = 0; j < 8; ++j) {
    int c = lane + 32 * j;
    Y[row * 256 + c] = f2bf((vals[j] - mu) * rsig * g[c] + be[c]);
  }
}

// ---------------------------------------------------------------------
// Generic bf16 WMMA GEMM: A(MxK bf16) @ Bw(KxN f32, rounded per tile).
// Block tile 64x128, 8 waves, each wave a 32x32 slab: 2 A-frags x 2 B-frags
// -> 4 WMMAs per 32-K step. grid = (M/64)*(N/128), block = 256
// ---------------------------------------------------------------------
template <bool HAS_BIAS, bool GELU, bool RES, bool OUT_BF16>
__global__ __launch_bounds__(256)
void gemm_bf16_kernel(const unsigned short* __restrict__ A,
                      const float* __restrict__ Bw,
                      const float* __restrict__ bias,
                      const float* __restrict__ res,
                      float* __restrict__ outF,
                      unsigned short* __restrict__ outB,
                      int M, int N, int K) {
  const int nb = N >> 7;
  const int bm = blockIdx.x / nb;
  const int bn = blockIdx.x % nb;
  const int tid = threadIdx.x;
  const int lane = tid & 31;
  const int wid = tid >> 5;
  const int wm = wid >> 2;   // 0..1 -> M offset 32*wm
  const int wn = wid & 3;    // 0..3 -> N offset 32*wn
  const int hi = lane >> 4, lm = lane & 15;

  __shared__ __align__(16) unsigned short As[64 * 32];   // [m][k]
  __shared__ __align__(16) unsigned short Bs[128 * 32];  // [n][k] (transposed)

  f32x8 acc00 = {}, acc01 = {}, acc10 = {}, acc11 = {};

  const int ar = tid >> 2;          // A tile row 0..63
  const int acg = (tid & 3) * 8;    // A tile col group
  const int bk = tid >> 3;          // B tile k 0..31
  const int bn16 = (tid & 7) * 16;  // B tile n group

  for (int k0 = 0; k0 < K; k0 += 32) {
    // ---- stage A tile (64x32 bf16, one 16B load per thread) ----
    const unsigned short* agp = A + (size_t)(bm * 64 + ar) * K + k0 + acg;
    *(bf16x8*)&As[ar * 32 + acg] = *(const bf16x8*)agp;
    if (k0 + 32 < K) __builtin_prefetch(agp + 32, 0, 1);  // global_prefetch_b8
    // ---- stage B tile (32x128 f32 -> bf16, transposed to [n][k]) ----
    const float* bgp = Bw + (size_t)(k0 + bk) * N + bn * 128 + bn16;
#pragma unroll
    for (int j = 0; j < 16; ++j) Bs[(bn16 + j) * 32 + bk] = f2bf(bgp[j]);
    __syncthreads();

    const unsigned short* ap0 = &As[(wm * 32 + lm) * 32];
    const unsigned short* ap1 = &As[(wm * 32 + 16 + lm) * 32];
    bf16x16 a0 = load_frag(ap0 + 8 * hi, ap0 + 16 + 8 * hi);
    bf16x16 a1 = load_frag(ap1 + 8 * hi, ap1 + 16 + 8 * hi);
    const unsigned short* bp0 = &Bs[(wn * 32 + lm) * 32 + 8 * hi];
    const unsigned short* bp1 = &Bs[(wn * 32 + 16 + lm) * 32 + 8 * hi];
    bf16x16 b0 = load_frag(bp0, bp0 + 16);
    bf16x16 b1 = load_frag(bp1, bp1 + 16);
    acc00 = wmma_bf16(a0, b0, acc00);
    acc01 = wmma_bf16(a0, b1, acc01);
    acc10 = wmma_bf16(a1, b0, acc10);
    acc11 = wmma_bf16(a1, b1, acc11);
    __syncthreads();
  }

  // ---- epilogue: C layout lane L, vgpr v -> (M = v+8*hi, N = lm) ----
#pragma unroll
  for (int v = 0; v < 8; ++v) {
    size_t r0 = (size_t)bm * 64 + wm * 32 + v + 8 * hi;
    size_t r1 = r0 + 16;
    int c0 = bn * 128 + wn * 32 + lm;
    float x00 = acc00[v], x01 = acc01[v], x10 = acc10[v], x11 = acc11[v];
    if (HAS_BIAS) {
      float bb0 = bias[c0], bb1 = bias[c0 + 16];
      x00 += bb0; x01 += bb1; x10 += bb0; x11 += bb1;
    }
    if (GELU) {
      x00 = 0.5f * x00 * (1.f + erff(x00 * 0.70710678118f));
      x01 = 0.5f * x01 * (1.f + erff(x01 * 0.70710678118f));
      x10 = 0.5f * x10 * (1.f + erff(x10 * 0.70710678118f));
      x11 = 0.5f * x11 * (1.f + erff(x11 * 0.70710678118f));
    }
    size_t o0 = r0 * N + c0, o1 = r1 * N + c0;
    if (RES) { x00 += res[o0]; x01 += res[o0 + 16]; x10 += res[o1]; x11 += res[o1 + 16]; }
    if (OUT_BF16) {
      outB[o0] = f2bf(x00); outB[o0 + 16] = f2bf(x01);
      outB[o1] = f2bf(x10); outB[o1 + 16] = f2bf(x11);
    } else {
      outF[o0] = x00; outF[o0 + 16] = x01;
      outF[o1] = x10; outF[o1 + 16] = x11;
    }
  }
}

// ---------------------------------------------------------------------
// Depthwise conv1d (k=3, pad 1, along f) + head split + L2-norm of q,k over d=32.
// grid = 65536, block = 256 (warp == head)
// ---------------------------------------------------------------------
__global__ __launch_bounds__(256)
void conv_heads_kernel(const unsigned short* __restrict__ qkv,
                       const float* __restrict__ dw,
                       unsigned short* __restrict__ qh,
                       unsigned short* __restrict__ kh,
                       unsigned short* __restrict__ vh) {
  int r = blockIdx.x;            // n*256 + f
  int n = r >> 8, f = r & 255;
  int c = threadIdx.x;
  int h = c >> 5, d = c & 31;
  size_t row = (size_t)r * 768;
  float o[3];
#pragma unroll
  for (int s = 0; s < 3; ++s) {
    int ch = s * 256 + c;
    float w0 = dw[ch * 3 + 0], w1 = dw[ch * 3 + 1], w2 = dw[ch * 3 + 2];
    float xm = (f > 0)   ? bf2f(qkv[row - 768 + ch]) : 0.f;
    float xc = bf2f(qkv[row + ch]);
    float xp = (f < 255) ? bf2f(qkv[row + 768 + ch]) : 0.f;
    o[s] = w0 * xm + w1 * xc + w2 * xp;
  }
  float q = o[0], k = o[1], v = o[2];
  float sq = q * q, sk = k * k;
#pragma unroll
  for (int m = 16; m >= 1; m >>= 1) { sq += __shfl_xor(sq, m, 32); sk += __shfl_xor(sk, m, 32); }
  float qn = q / fmaxf(sqrtf(sq), 1e-12f);
  float kn = k / fmaxf(sqrtf(sk), 1e-12f);
  size_t oidx = (((size_t)(n * 8 + h)) * 256 + f) * 32 + d;
  qh[oidx] = f2bf(qn);
  kh[oidx] = f2bf(kn);
  vh[oidx] = f2bf(v);
}

// ---------------------------------------------------------------------
// Attention per (n,h), two-pass: S = q k^T * temp fully in registers
// (16 rows x 256 cols per wave = 128 f32 VGPRs/lane), one softmax, then
// P@V. q,k tiles fetched by the Tensor Data Mover (TDM) when available.
// grid = N*H = 2048, block = 512 (16 waves, wave w owns 16 q-rows)
// ---------------------------------------------------------------------
__global__ __launch_bounds__(512)
void attention_kernel(const unsigned short* __restrict__ qh,
                      const unsigned short* __restrict__ kh,
                      const unsigned short* __restrict__ vh,
                      const float* __restrict__ temp,
                      unsigned short* __restrict__ ao) {
  int nh = blockIdx.x;
  int n = nh >> 3, h = nh & 7;
  const float tscale = temp[h];
  const size_t base = (size_t)nh * 8192;

  __shared__ __align__(16) unsigned short qs[256 * 32];    // [f][d]
  __shared__ __align__(16) unsigned short ks[256 * 32];    // [t][d]
  __shared__ __align__(16) unsigned short vts[32 * 256];   // [d][t]
  __shared__ __align__(16) unsigned short ps[16][16 * 32]; // per-wave P stage

#if HAVE_TDM
  if (threadIdx.x < 32) {            // one wave issues both TDM descriptors
    tdm_load_b16((unsigned)(size_t)(void*)qs, qh + base, 8192);
    tdm_load_b16((unsigned)(size_t)(void*)ks, kh + base, 8192);
    __builtin_amdgcn_s_wait_tensorcnt(0);
  }
  for (int i = threadIdx.x; i < 8192; i += 512)
    vts[(i & 31) * 256 + (i >> 5)] = vh[base + i];
#else
  for (int i = threadIdx.x; i < 8192; i += 512) {
    qs[i] = qh[base + i];
    ks[i] = kh[base + i];
    vts[(i & 31) * 256 + (i >> 5)] = vh[base + i];
  }
#endif
  __syncthreads();

  int w = threadIdx.x >> 5, lane = threadIdx.x & 31;
  int hi = lane >> 4, lm = lane & 15;
  int m0 = w * 16;

  const unsigned short* qp = &qs[(m0 + lm) * 32];
  bf16x16 qA = load_frag(qp + 8 * hi, qp + 16 + 8 * hi);

  // ---- pass 1: all of S for this wave's 16 rows ----
  f32x8 sreg[16];
#pragma unroll
  for (int c = 0; c < 8; ++c) {
    int t0 = c * 32;
    const unsigned short* kp0 = &ks[(t0 + lm) * 32 + 8 * hi];
    const unsigned short* kp1 = &ks[(t0 + 16 + lm) * 32 + 8 * hi];
    bf16x16 kB0 = load_frag(kp0, kp0 + 16);
    bf16x16 kB1 = load_frag(kp1, kp1 + 16);
    f32x8 z = {};
    f32x8 u0 = wmma_bf16(qA, kB0, z);
    f32x8 u1 = wmma_bf16(qA, kB1, z);
#pragma unroll
    for (int e = 0; e < 8; ++e) { u0[e] *= tscale; u1[e] *= tscale; }
    sreg[2 * c] = u0;
    sreg[2 * c + 1] = u1;
  }

  // ---- softmax: rows live in 16-lane halves -> butterfly xor 1,2,4,8 ----
  float linv[8];
#pragma unroll
  for (int v = 0; v < 8; ++v) {
    float mx = sreg[0][v];
#pragma unroll
    for (int i = 1; i < 16; ++i) mx = fmaxf(mx, sreg[i][v]);
    mx = fmaxf(mx, __shfl_xor(mx, 1, 32));
    mx = fmaxf(mx, __shfl_xor(mx, 2, 32));
    mx = fmaxf(mx, __shfl_xor(mx, 4, 32));
    mx = fmaxf(mx, __shfl_xor(mx, 8, 32));
    float sum = 0.f;
#pragma unroll
    for (int i = 0; i < 16; ++i) {
      float p = __expf(sreg[i][v] - mx);
      sreg[i][v] = p;
      sum += p;
    }
    sum += __shfl_xor(sum, 1, 32);
    sum += __shfl_xor(sum, 2, 32);
    sum += __shfl_xor(sum, 4, 32);
    sum += __shfl_xor(sum, 8, 32);
    linv[v] = 1.f / sum;
  }

  // ---- pass 2: O = (P/l) @ V, staging P chunks via wave-local LDS ----
  f32x8 o0 = {}, o1 = {};
  unsigned short* myp = &ps[w][0];
#pragma unroll
  for (int c = 0; c < 8; ++c) {
    int t0 = c * 32;
#pragma unroll
    for (int v = 0; v < 8; ++v) {
      int M = v + 8 * hi;              // C-layout -> A-layout via LDS
      myp[M * 32 + lm] = f2bf(sreg[2 * c][v] * linv[v]);
      myp[M * 32 + 16 + lm] = f2bf(sreg[2 * c + 1][v] * linv[v]);
    }
    const unsigned short* pp = &myp[lm * 32];
    bf16x16 pA = load_frag(pp + 8 * hi, pp + 16 + 8 * hi);
    const unsigned short* vp0 = &vts[lm * 256 + t0 + 8 * hi];
    const unsigned short* vp1 = &vts[(16 + lm) * 256 + t0 + 8 * hi];
    bf16x16 vB0 = load_frag(vp0, vp0 + 16);
    bf16x16 vB1 = load_frag(vp1, vp1 + 16);
    o0 = wmma_bf16(pA, vB0, o0);
    o1 = wmma_bf16(pA, vB1, o1);
  }

#pragma unroll
  for (int v = 0; v < 8; ++v) {
    int fr = m0 + v + 8 * hi;
    size_t oi = ((size_t)n * 256 + fr) * 256 + h * 32;
    ao[oi + lm] = f2bf(o0[v]);
    ao[oi + 16 + lm] = f2bf(o1[v]);
  }
}

// ---------------------------------------------------------------------
// unpack y2[(b*T+t)*F + f][c] f32 -> out (B,C,T,F) f32
// ---------------------------------------------------------------------
__global__ __launch_bounds__(256)
void unpack_out_kernel(const float* __restrict__ y2, float* __restrict__ out) {
  int blk = blockIdx.x;
  int ftile = blk & 7, ctile = (blk >> 3) & 7, bt = blk >> 6;
  int b = bt >> 6, t = bt & 63;
  __shared__ float tile[32][33];
  int ti = threadIdx.x & 31, to = threadIdx.x >> 5;
#pragma unroll
  for (int k = 0; k < 4; ++k) {
    int ff = to + 8 * k;
    tile[ff][ti] = y2[((size_t)bt * 256 + ftile * 32 + ff) * 256 + ctile * 32 + ti];
  }
  __syncthreads();
#pragma unroll
  for (int k = 0; k < 4; ++k) {
    int cc = to + 8 * k;
    out[(((size_t)(b * 256 + ctile * 32 + cc)) * 64 + t) * 256 + ftile * 32 + ti] =
        tile[ti][cc];
  }
}

// =====================================================================
extern "C" void kernel_launch(void* const* d_in, const int* in_sizes, int n_in,
                              void* d_out, int out_size, void* d_ws, size_t ws_size,
                              hipStream_t stream) {
  (void)in_sizes; (void)n_in; (void)out_size; (void)ws_size;
  const float* x     = (const float*)d_in[0];
  const float* n1g   = (const float*)d_in[1];
  const float* n1b   = (const float*)d_in[2];
  const float* Wqkv  = (const float*)d_in[3];
  const float* dw    = (const float*)d_in[4];
  const float* Wproj = (const float*)d_in[5];
  const float* temp  = (const float*)d_in[6];
  const float* n2g   = (const float*)d_in[7];
  const float* n2b   = (const float*)d_in[8];
  const float* W1    = (const float*)d_in[9];
  const float* b1    = (const float*)d_in[10];
  const float* W2    = (const float*)d_in[11];
  const float* b2    = (const float*)d_in[12];
  float* out = (float*)d_out;

  char* ws = (char*)d_ws;
  const size_t MB = 1ull << 20;
  float*          xs  = (float*)(ws + 0);                   // 64MB
  unsigned short* xn  = (unsigned short*)(ws + 64 * MB);    // 32MB
  unsigned short* qkv = (unsigned short*)(ws + 96 * MB);    // 96MB
  unsigned short* qh  = (unsigned short*)(ws + 192 * MB);   // 32MB
  unsigned short* kh  = (unsigned short*)(ws + 224 * MB);   // 32MB
  unsigned short* vh  = (unsigned short*)(ws + 256 * MB);   // 32MB
  unsigned short* ao  = (unsigned short*)(ws + 64 * MB);    // reuse xn
  float*          y   = (float*)(ws + 96 * MB);             // reuse qkv lo
  unsigned short* yn  = (unsigned short*)(ws + 160 * MB);   // reuse qkv hi
  unsigned short* h1  = (unsigned short*)(ws + 192 * MB);   // reuse qh/kh
  float*          y2  = (float*)(ws + 0);                   // reuse xs

  const int M = 65536;  // B*T*F rows

  pack_xs_kernel<<<16384, 256, 0, stream>>>(x, xs);
  layernorm_kernel<<<M / 8, 256, 0, stream>>>(xs, n1g, n1b, xn);
  gemm_bf16_kernel<false, false, false, true>
      <<<(M / 64) * 6, 256, 0, stream>>>(xn, Wqkv, nullptr, nullptr,
                                         nullptr, qkv, M, 768, 256);
  conv_heads_kernel<<<M, 256, 0, stream>>>(qkv, dw, qh, kh, vh);
  attention_kernel<<<2048, 512, 0, stream>>>(qh, kh, vh, temp, ao);
  gemm_bf16_kernel<false, false, true, false>
      <<<(M / 64) * 2, 256, 0, stream>>>(ao, Wproj, nullptr, xs,
                                         y, nullptr, M, 256, 256);
  layernorm_kernel<<<M / 8, 256, 0, stream>>>(y, n2g, n2b, yn);
  gemm_bf16_kernel<true, true, false, true>
      <<<(M / 64) * 4, 256, 0, stream>>>(yn, W1, b1, nullptr,
                                         nullptr, h1, M, 512, 256);
  gemm_bf16_kernel<true, false, true, false>
      <<<(M / 64) * 2, 256, 0, stream>>>(h1, W2, b2, y,
                                         y2, nullptr, M, 256, 512);
  unpack_out_kernel<<<16384, 256, 0, stream>>>(y2, out);
}
